// EarthMoversDistance_56831007261025
// MI455X (gfx1250) — compile-verified, matrix operand order checked by reading, not run on hardware
//
#include <hip/hip_runtime.h>
#include <hip/hip_bf16.h>

// ---------------------------------------------------------------------------
// Sinkhorn EMD on MI455X (gfx1250).
//   K, KT materialized once in bf16 (134 MB total -> fits 192 MB L2).
//   100 iterations x 2 GEMV passes, each GEMV uses v_wmma_f32_16x16x32_bf16
//   with the vector broadcast into all 16 columns of the B operand.
// ---------------------------------------------------------------------------

typedef __attribute__((ext_vector_type(16))) __bf16 v16bf;
typedef __attribute__((ext_vector_type(8)))  float  v8f;

#define BATCH 8
#define NPTS  2048   // N == M == 2048 (power of two, 2^11)
#define EPS_DIV 1e-8f
#define MAX_ITER 100

union ABPack { v16bf v; uint4 q[2]; };

// ---------------------------------------------------------------------------
// Kernel 1: build K[b][n][m] and KT[b][m][n] in bf16.
// block = (16,16); LDS staging so the transposed store is also coalesced.
// ---------------------------------------------------------------------------
__global__ __launch_bounds__(256)
void build_K_kernel(const float* __restrict__ src, const float* __restrict__ tgt,
                    __bf16* __restrict__ Kq, __bf16* __restrict__ KTq)
{
    __shared__ float tile[16][17];
    const int b  = blockIdx.z;
    const int nb = blockIdx.y;
    const int mb = blockIdx.x;
    const int tx = threadIdx.x;
    const int ty = threadIdx.y;

    const int n = nb * 16 + ty;
    const int m = mb * 16 + tx;

    const float* s = src + ((size_t)b * NPTS + n) * 3;
    const float* t = tgt + ((size_t)b * NPTS + m) * 3;
    const float dx = s[0] - t[0];
    const float dy = s[1] - t[1];
    const float dz = s[2] - t[2];
    const float c  = sqrtf(dx * dx + dy * dy + dz * dz);
    // K = exp(-min(c/0.01, 50));  exp(-50)=1.9e-22: representable in bf16
    // (fp32 exponent range), flushed to zero by fp16/fp8 -> bf16 mandatory.
    const float k  = __expf(-fminf(c * 100.0f, 50.0f));

    Kq[((size_t)b * NPTS + n) * NPTS + m] = (__bf16)k;

    tile[ty][tx] = k;
    __syncthreads();
    // KT[b][mb*16+ty][nb*16+tx] = K[b][nb*16+tx][mb*16+ty]
    KTq[((size_t)b * NPTS + (mb * 16 + ty)) * NPTS + (nb * 16 + tx)] =
        (__bf16)tile[tx][ty];
}

// ---------------------------------------------------------------------------
// Kernel 2: u/v init to 1.0 (float + bf16 mirrors).
// ---------------------------------------------------------------------------
__global__ __launch_bounds__(256)
void init_uv_kernel(float* __restrict__ u, float* __restrict__ v,
                    __bf16* __restrict__ ubf, __bf16* __restrict__ vbf)
{
    const int i = blockIdx.x * blockDim.x + threadIdx.x;
    if (i < BATCH * NPTS) {
        u[i]   = 1.0f;
        v[i]   = 1.0f;
        ubf[i] = (__bf16)1.0f;
        vbf[i] = (__bf16)1.0f;
    }
}

// ---------------------------------------------------------------------------
// Kernel 3: one Sinkhorn half-step:  out = rho / max(Mat * in, eps)
// Mat is row-major bf16 (K for the u-update, KT for the v-update).
// One wave handles 16 rows; 64 x v_wmma_f32_16x16x32_bf16 over k=2048.
// A operand layout (16-bit 16x32, ISA 7.12.2): lane L<16 -> row L,
// k in {base+0..7, base+16..23}; lane L>=16 -> row L-16, k in {8..15,24..31}.
// B operand: every column j gets B[k][j] = in[k]  (broadcast vector), so
// every column of D equals Mat*in; D rows striped over the 8 acc VGPRs.
// ---------------------------------------------------------------------------
__global__ __launch_bounds__(128)
void sinkhorn_gemv_kernel(const __bf16* __restrict__ Mat,
                          const __bf16* __restrict__ in_bf,
                          float* __restrict__ out_f,
                          __bf16* __restrict__ out_bf,
                          float rho)
{
    const int lane = threadIdx.x & 31;
    const int wave = blockIdx.x * (blockDim.x >> 5) + (threadIdx.x >> 5);
    const int b    = wave >> 7;          // 128 row-tiles per batch
    const int tile = wave & 127;
    const int row  = tile * 16 + (lane & 15);
    const int koff = (lane >> 4) * 8;    // 0 for lanes 0-15, 8 for lanes 16-31

    const __bf16* arow = Mat + ((size_t)b * NPTS + row) * NPTS;
    const __bf16* vvec = in_bf + b * NPTS;

    v8f acc = {0.f, 0.f, 0.f, 0.f, 0.f, 0.f, 0.f, 0.f};

    for (int kk = 0; kk < NPTS; kk += 32) {
        ABPack A, Bv;
        A.q[0]  = *(const uint4*)(arow + kk + koff);        // k: koff+0..7
        A.q[1]  = *(const uint4*)(arow + kk + koff + 16);   // k: koff+16..23
        Bv.q[0] = *(const uint4*)(vvec + kk + koff);        // broadcast vector
        Bv.q[1] = *(const uint4*)(vvec + kk + koff + 16);
        acc = __builtin_amdgcn_wmma_f32_16x16x32_bf16(
            /*neg_a=*/false, A.v, /*neg_b=*/false, Bv.v,
            /*c_mod=*/(short)0, acc, /*reuse_a=*/false, /*reuse_b=*/false);
    }

    // D layout: VGPR r = row (tile*16 + r) on lanes 0-15, row (tile*16+8+r)
    // on lanes 16-31; all 16 columns identical -> lanes 0 and 16 write.
    if ((lane & 15) == 0) {
        const int rbase = tile * 16 + koff;
        float*  o  = out_f  + (size_t)b * NPTS + rbase;
        __bf16* ob = out_bf + (size_t)b * NPTS + rbase;
#pragma unroll
        for (int r = 0; r < 8; ++r) {
            const float val = rho / fmaxf(acc[r], EPS_DIV);
            o[r]  = val;
            ob[r] = (__bf16)val;
        }
    }
}

// ---------------------------------------------------------------------------
// Kernel 4: partial sums of  u_n * K_nm * v_m * cost_nm  (cost recomputed
// from raw points). Deterministic fixed-tree block reduction.
// ---------------------------------------------------------------------------
#define EMD_ITERS 16
__global__ __launch_bounds__(256)
void emd_partial_kernel(const float* __restrict__ src, const float* __restrict__ tgt,
                        const __bf16* __restrict__ Kq,
                        const float* __restrict__ u, const float* __restrict__ v,
                        float* __restrict__ partials)
{
    const int tid = threadIdx.x;
    float sum = 0.0f;
    size_t idx = (size_t)blockIdx.x * (blockDim.x * EMD_ITERS) + tid;
#pragma unroll 4
    for (int i = 0; i < EMD_ITERS; ++i, idx += 256) {
        const int m = (int)(idx & (NPTS - 1));
        const int n = (int)((idx >> 11) & (NPTS - 1));
        const int b = (int)(idx >> 22);
        const float* s = src + ((size_t)b * NPTS + n) * 3;
        const float* t = tgt + ((size_t)b * NPTS + m) * 3;
        const float dx = s[0] - t[0];
        const float dy = s[1] - t[1];
        const float dz = s[2] - t[2];
        const float c  = sqrtf(dx * dx + dy * dy + dz * dz);
        const float k  = (float)Kq[idx];
        sum += u[b * NPTS + n] * k * v[b * NPTS + m] * c;
    }
    __shared__ float red[256];
    red[tid] = sum;
    __syncthreads();
#pragma unroll
    for (int s = 128; s > 0; s >>= 1) {
        if (tid < s) red[tid] += red[tid + s];
        __syncthreads();
    }
    if (tid == 0) partials[blockIdx.x] = red[0];
}

__global__ __launch_bounds__(1024)
void emd_reduce_kernel(const float* __restrict__ partials, float* __restrict__ out,
                       int nparts)
{
    __shared__ float red[1024];
    float s = 0.0f;
    for (int i = threadIdx.x; i < nparts; i += 1024) s += partials[i];
    red[threadIdx.x] = s;
    __syncthreads();
#pragma unroll
    for (int st = 512; st > 0; st >>= 1) {
        if (threadIdx.x < st) red[threadIdx.x] += red[threadIdx.x + st];
        __syncthreads();
    }
    if (threadIdx.x == 0) out[0] = red[0] / (float)BATCH;  // mean over batches
}

// ---------------------------------------------------------------------------
extern "C" void kernel_launch(void* const* d_in, const int* in_sizes, int n_in,
                              void* d_out, int out_size, void* d_ws, size_t ws_size,
                              hipStream_t stream)
{
    (void)in_sizes; (void)n_in; (void)out_size; (void)ws_size;
    const float* src = (const float*)d_in[0];   // (8, 2048, 3) f32
    const float* tgt = (const float*)d_in[1];   // (8, 2048, 3) f32
    float* out = (float*)d_out;                 // scalar f32

    // Workspace layout
    const size_t kmat_bytes = (size_t)BATCH * NPTS * NPTS * sizeof(__bf16); // 67 MB
    const size_t vec_f      = (size_t)BATCH * NPTS * sizeof(float);
    const size_t vec_bf     = (size_t)BATCH * NPTS * sizeof(__bf16);
    char* ws = (char*)d_ws;
    size_t off = 0;
    __bf16* Kq  = (__bf16*)(ws + off); off += kmat_bytes;
    __bf16* KTq = (__bf16*)(ws + off); off += kmat_bytes;
    float*  u   = (float*)(ws + off);  off += vec_f;
    float*  v   = (float*)(ws + off);  off += vec_f;
    __bf16* ubf = (__bf16*)(ws + off); off += vec_bf;
    __bf16* vbf = (__bf16*)(ws + off); off += vec_bf;
    float* partials = (float*)(ws + off);

    // 1) Build K and K^T in bf16
    build_K_kernel<<<dim3(NPTS / 16, NPTS / 16, BATCH), dim3(16, 16), 0, stream>>>(
        src, tgt, Kq, KTq);

    // 2) u = v = 1
    init_uv_kernel<<<(BATCH * NPTS + 255) / 256, 256, 0, stream>>>(u, v, ubf, vbf);

    // 3) 100 Sinkhorn iterations; each half-step is a WMMA-bf16 GEMV pass.
    //    waves = 8 batches * 128 row-tiles = 1024; block = 128 thr (4 waves).
    const int gemv_blocks = (BATCH * (NPTS / 16)) / 4;  // 256
    const float mu_val = 1.0f / (float)NPTS;
    const float nu_val = 1.0f / (float)NPTS;
    for (int it = 0; it < MAX_ITER; ++it) {
        sinkhorn_gemv_kernel<<<gemv_blocks, 128, 0, stream>>>(Kq,  vbf, u, ubf, mu_val);
        sinkhorn_gemv_kernel<<<gemv_blocks, 128, 0, stream>>>(KTq, ubf, v, vbf, nu_val);
    }

    // 4) emd = mean_b sum_{n,m} u_n K_nm v_m cost_nm
    const int nentries_log2 = 25;                       // 8*2048*2048 = 2^25
    const int nblocks = 1 << (nentries_log2 - 12);      // 256 thr * 16 iters = 2^12
    emd_partial_kernel<<<nblocks, 256, 0, stream>>>(src, tgt, Kq, u, v, partials);
    emd_reduce_kernel<<<1, 1024, 0, stream>>>(partials, out, nblocks);
}
